// TrackMamba_23819888623735
// MI455X (gfx1250) — matrix operand order repaired and checked by reading, not compiled, not run on hardware
//
#include <hip/hip_runtime.h>
#include <hip/hip_bf16.h>
#include <math.h>

// ---------------- model constants (from reference) ----------------
#define VV   8
#define DD   256
#define LL   4096
#define BB   2
#define NLY  4
#define DI   512
#define DS   64
#define DC   4
#define HD   64
#define NH   (DI / HD)          // 8
#define CD   (DI + 2 * DS)      // 640
#define DIP  (2 * DI + 2 * DS + NH) // 1160
#define BL   (BB * LL)          // 8192

// ---------------- vector types for WMMA ----------------
typedef float          v8f   __attribute__((ext_vector_type(8)));
typedef __bf16         v8bf  __attribute__((ext_vector_type(8)));
typedef __bf16         v16bf __attribute__((ext_vector_type(16)));
typedef unsigned short v8u   __attribute__((ext_vector_type(8)));
typedef unsigned short v16u  __attribute__((ext_vector_type(16)));

union Frag16 {
    v16u  u;
    v8u   h[2];
    v16bf bf;
};

__device__ inline float siluf(float x) { return x / (1.f + __expf(-x)); }

// packed fp32 -> bf16 convert + 16B LDS store (backend emits v_cvt_pk_bf16_f32)
__device__ inline void cvt_store8(unsigned short* dst, float4 x0, float4 x1) {
    v8f f;
    f[0] = x0.x; f[1] = x0.y; f[2] = x0.z; f[3] = x0.w;
    f[4] = x1.x; f[5] = x1.y; f[6] = x1.z; f[7] = x1.w;
    v8bf b = __builtin_convertvector(f, v8bf);
    *(v8u*)dst = __builtin_bit_cast(v8u, b);
}

// ---------------- block-wide all-reduce (blockDim.x == 256, wave32) ------
__device__ inline float blockAllReduceSum(float v, float* sm /* >= 9 floats */) {
    int lane = threadIdx.x & 31, wid = threadIdx.x >> 5;
    #pragma unroll
    for (int o = 16; o > 0; o >>= 1) v += __shfl_xor(v, o, 32);
    if (lane == 0) sm[wid] = v;
    __syncthreads();
    if (threadIdx.x == 0) {
        float s = 0.f;
        #pragma unroll
        for (int w = 0; w < 8; w++) s += sm[w];
        sm[8] = s;
    }
    __syncthreads();
    float r = sm[8];
    __syncthreads();
    return r;
}

// =====================================================================
// GEMM: C[M,N] = A[M,K] @ W[N,K]^T  (+bias, optional SiLU, +residual)
// bf16 WMMA, 64x128 block tile, K-step 32, 8 waves (wave32),
// double-buffered LDS + software pipeline:
//   gload(next) -> ds_load all frags(cur) -> 4x wmma -> cvt+ds_store(next)
// All four B fragments live in distinct registers so the ds_loads clause
// up front and drain with staged s_wait_dscnt between WMMAs.
// =====================================================================
#define LDT 40   // LDS half-stride (32 + 8 pad), 80B rows -> 16B aligned

__global__ __launch_bounds__(256)
void k_gemm(const float* __restrict__ A, const float* __restrict__ W,
            float* __restrict__ C, const float* __restrict__ bias,
            const float* __restrict__ res, int M, int N, int K, int act) {
    __shared__ __align__(16) unsigned short ldsA[2][64 * LDT];
    __shared__ __align__(16) unsigned short ldsB[2][128 * LDT];

    const int tid  = threadIdx.x;
    const int m0   = blockIdx.y * 64;
    const int n0   = blockIdx.x * 128;
    const int wid  = tid >> 5;
    const int lane = tid & 31;
    const int m_off = (wid >> 1) * 16;   // 0,16,32,48
    const int n_off = (wid & 1) * 64;    // 0,64  (wave strip: 16M x 64N)

    // staging coords: A 8 fp32/thread, B 16 fp32/thread
    const int arow = tid >> 2;           // 0..63
    const int acol = (tid & 3) * 8;      // 0,8,16,24
    const int brow = tid >> 1;           // 0..127
    const int bcol = (tid & 1) * 16;     // 0,16

    v8f acc0, acc1, acc2, acc3;
    #pragma unroll
    for (int i = 0; i < 8; i++) { acc0[i] = 0.f; acc1[i] = 0.f; acc2[i] = 0.f; acc3[i] = 0.f; }

    // -------- prologue: stage K-tile 0 into buffer 0 --------
    float4 a0, a1, bv0, bv1, bv2, bv3;
    {
        const float* gp = A + (size_t)(m0 + arow) * K + acol;
        a0 = *(const float4*)gp;
        a1 = *(const float4*)(gp + 4);
        int gn = n0 + brow;
        if (gn < N) {
            const float* wp = W + (size_t)gn * K + bcol;
            bv0 = *(const float4*)wp;
            bv1 = *(const float4*)(wp + 4);
            bv2 = *(const float4*)(wp + 8);
            bv3 = *(const float4*)(wp + 12);
        } else {
            bv0 = bv1 = bv2 = bv3 = make_float4(0.f, 0.f, 0.f, 0.f);
        }
        cvt_store8(&ldsA[0][arow * LDT + acol], a0, a1);
        cvt_store8(&ldsB[0][brow * LDT + bcol], bv0, bv1);
        cvt_store8(&ldsB[0][brow * LDT + bcol + 8], bv2, bv3);
    }
    __syncthreads();

    for (int kk = 0; kk < K; kk += 32) {
        const int  cur     = (kk >> 5) & 1;
        const int  nxt     = cur ^ 1;
        const bool hasNext = (kk + 32) < K;

        // ---- issue next tile's global loads (latency hidden by wmma) ----
        if (hasNext) {
            const float* gp = A + (size_t)(m0 + arow) * K + (kk + 32) + acol;
            a0 = *(const float4*)gp;
            a1 = *(const float4*)(gp + 4);
            if (kk + 64 < K) __builtin_prefetch(gp + 32, 0, 1);  // global_prefetch_b8
            int gn = n0 + brow;
            if (gn < N) {
                const float* wp = W + (size_t)gn * K + (kk + 32) + bcol;
                bv0 = *(const float4*)wp;
                bv1 = *(const float4*)(wp + 4);
                bv2 = *(const float4*)(wp + 8);
                bv3 = *(const float4*)(wp + 12);
                if (kk + 64 < K) __builtin_prefetch(wp + 32, 0, 1);
            } else {
                bv0 = bv1 = bv2 = bv3 = make_float4(0.f, 0.f, 0.f, 0.f);
            }
        }

        // ---- load ALL fragments from current buffer first ----
        const unsigned short* lA = ldsA[cur];
        const unsigned short* lB = ldsB[cur];
        const int rA = m_off + (lane & 15);
        const int kb = (lane >> 4) * 8;
        const int kh = (lane >> 4) * 16;                     // K 0..15 / 16..31
        const int cB = n_off + (lane & 15);

        Frag16 af, bf0, bf1, bf2, bf3;
        af.h[0]  = *(const v8u*)&lA[rA * LDT + kh];          // K = kh..kh+7
        af.h[1]  = *(const v8u*)&lA[rA * LDT + kh + 8];      // K = kh+8..kh+15
        bf0.h[0] = *(const v8u*)&lB[(cB     ) * LDT + kh];
        bf0.h[1] = *(const v8u*)&lB[(cB     ) * LDT + kh + 8];
        bf1.h[0] = *(const v8u*)&lB[(cB + 16) * LDT + kh];
        bf1.h[1] = *(const v8u*)&lB[(cB + 16) * LDT + kh + 8];
        bf2.h[0] = *(const v8u*)&lB[(cB + 32) * LDT + kh];
        bf2.h[1] = *(const v8u*)&lB[(cB + 32) * LDT + kh + 8];
        bf3.h[0] = *(const v8u*)&lB[(cB + 48) * LDT + kh];
        bf3.h[1] = *(const v8u*)&lB[(cB + 48) * LDT + kh + 8];

        acc0 = __builtin_amdgcn_wmma_f32_16x16x32_bf16(false, af.bf, false, bf0.bf,
                                                       (short)0, acc0, false, false);
        acc1 = __builtin_amdgcn_wmma_f32_16x16x32_bf16(false, af.bf, false, bf1.bf,
                                                       (short)0, acc1, false, false);
        acc2 = __builtin_amdgcn_wmma_f32_16x16x32_bf16(false, af.bf, false, bf2.bf,
                                                       (short)0, acc2, false, false);
        acc3 = __builtin_amdgcn_wmma_f32_16x16x32_bf16(false, af.bf, false, bf3.bf,
                                                       (short)0, acc3, false, false);

        // ---- convert + store next tile behind the WMMAs ----
        if (hasNext) {
            cvt_store8(&ldsA[nxt][arow * LDT + acol], a0, a1);
            cvt_store8(&ldsB[nxt][brow * LDT + bcol], bv0, bv1);
            cvt_store8(&ldsB[nxt][brow * LDT + bcol + 8], bv2, bv3);
        }
        __syncthreads();
    }

    // ---- epilogue + store per 32-bit C/D layout ----
    const int colN = lane & 15;
    const int rsel = lane >> 4;
    v8f accs[4] = { acc0, acc1, acc2, acc3 };
    #pragma unroll
    for (int j = 0; j < 4; j++) {
        int gn = n0 + n_off + j * 16 + colN;
        if (gn < N) {
            #pragma unroll
            for (int i = 0; i < 8; i++) {
                int gm = m0 + m_off + i + rsel * 8;
                size_t idx = (size_t)gm * N + gn;
                float v = accs[j][i];
                if (bias) v += bias[gn];
                if (act)  v = siluf(v);
                if (res)  v += res[idx];
                C[idx] = v;
            }
        }
    }
}

// ================= embedding =================
__global__ __launch_bounds__(256)
void k_embed(const int* __restrict__ x, const float* __restrict__ emb,
             const float* __restrict__ pos, float* __restrict__ h) {
    int r = blockIdx.x, d = threadIdx.x;
    int tok = x[r];
    int l = r % LL;
    h[(size_t)r * DD + d] = emb[(size_t)tok * DD + d] + pos[(size_t)l * DD + d];
}

// ================= depthwise causal conv + SiLU, softplus(dt) =========
__global__ __launch_bounds__(256)
void k_conv(const float* __restrict__ zx, const float* __restrict__ cw,
            const float* __restrict__ cb, const float* __restrict__ dtb,
            float* __restrict__ xbc_out, float* __restrict__ dt_out) {
    int r = blockIdx.x;          // b*L + l
    int l = r % LL;
    for (int c = threadIdx.x; c < CD; c += blockDim.x) {
        float acc = cb[c];
        #pragma unroll
        for (int k = 0; k < DC; k++) {
            int ll = l - (DC - 1) + k;
            if (ll >= 0)
                acc += zx[(size_t)(r - (DC - 1) + k) * DIP + DI + c] * cw[c * DC + k];
        }
        xbc_out[(size_t)r * CD + c] = siluf(acc);
    }
    if (threadIdx.x < NH) {
        float v = zx[(size_t)r * DIP + DI + CD + threadIdx.x] + dtb[threadIdx.x];
        dt_out[(size_t)r * NH + threadIdx.x] = (v > 20.f) ? v : log1pf(__expf(v));
    }
}

// ================= SSM recurrence: one block per (b, head) ============
// thread (p = tid>>2, q = tid&3) owns state h[p][q*16 .. q*16+15] in regs.
__global__ __launch_bounds__(256)
void k_scan(const float* __restrict__ xbc, const float* __restrict__ dt,
            const float* __restrict__ A_log, const float* __restrict__ Dp,
            float* __restrict__ yout) {
    const int b  = blockIdx.x / NH;
    const int hh = blockIdx.x % NH;
    const int p  = threadIdx.x >> 2;
    const int q  = threadIdx.x & 3;
    const float Av = -__expf(A_log[hh]);
    const float Dh = Dp[hh];

    float s[16];
    #pragma unroll
    for (int j = 0; j < 16; j++) s[j] = 0.f;

    for (int t = 0; t < LL; t++) {
        size_t r   = (size_t)b * LL + t;
        float dtv  = dt[r * NH + hh];
        float dA   = __expf(dtv * Av);
        float xp   = xbc[r * CD + hh * HD + p];
        float dxp  = dtv * xp;
        const float* Bp = xbc + r * CD + DI + q * 16;
        const float* Cp = Bp + DS;
        float acc = 0.f;
        #pragma unroll
        for (int j4 = 0; j4 < 4; j4++) {
            float4 Bv = *(const float4*)(Bp + 4 * j4);
            float4 Cv = *(const float4*)(Cp + 4 * j4);
            s[4*j4+0] = s[4*j4+0] * dA + dxp * Bv.x; acc += s[4*j4+0] * Cv.x;
            s[4*j4+1] = s[4*j4+1] * dA + dxp * Bv.y; acc += s[4*j4+1] * Cv.y;
            s[4*j4+2] = s[4*j4+2] * dA + dxp * Bv.z; acc += s[4*j4+2] * Cv.z;
            s[4*j4+3] = s[4*j4+3] * dA + dxp * Bv.w; acc += s[4*j4+3] * Cv.w;
        }
        acc += __shfl_xor(acc, 1, 32);   // reduce over q (same wave, wave32)
        acc += __shfl_xor(acc, 2, 32);
        if (q == 0)
            yout[r * DI + hh * HD + p] = acc + Dh * xp;
    }
}

// ================= gated RMS norm (in-place on y) =====================
__global__ __launch_bounds__(256)
void k_gated_norm(float* __restrict__ y, const float* __restrict__ zx,
                  const float* __restrict__ nw) {
    __shared__ float sm[9];
    int r = blockIdx.x, d = threadIdx.x;
    float z0 = zx[(size_t)r * DIP + d];
    float z1 = zx[(size_t)r * DIP + d + 256];
    float y0 = y[(size_t)r * DI + d]       * siluf(z0);
    float y1 = y[(size_t)r * DI + d + 256] * siluf(z1);
    float ss = blockAllReduceSum(y0 * y0 + y1 * y1, sm) * (1.f / DI);
    float sc = rsqrtf(ss + 1e-5f);
    y[(size_t)r * DI + d]       = y0 * sc * nw[d];
    y[(size_t)r * DI + d + 256] = y1 * sc * nw[d + 256];
}

// ================= LayerNorm over D=256 ===============================
__global__ __launch_bounds__(256)
void k_layernorm(const float* __restrict__ in, const float* __restrict__ w,
                 const float* __restrict__ b, float* __restrict__ out) {
    __shared__ float sm[9];
    int r = blockIdx.x, d = threadIdx.x;
    float v   = in[(size_t)r * DD + d];
    float mu  = blockAllReduceSum(v, sm) * (1.f / DD);
    float dv  = v - mu;
    float var = blockAllReduceSum(dv * dv, sm) * (1.f / DD);
    out[(size_t)r * DD + d] = dv * rsqrtf(var + 1e-5f) * w[d] + b[d];
}

// ================= output heads =======================================
__global__ __launch_bounds__(256)
void k_heads(const float* __restrict__ h, const float* __restrict__ trw,
             const float* __restrict__ trb, const float* __restrict__ clw,
             const float* __restrict__ clb, float* __restrict__ out) {
    __shared__ float sm[9];
    int r = blockIdx.x, d = threadIdx.x;
    float hv = h[(size_t)r * DD + d];
    float s1 = blockAllReduceSum(hv * trw[d], sm);
    float s2 = blockAllReduceSum(hv * clw[d], sm);
    if (d == 0) {
        out[r]      = s1 + trb[0];
        out[BL + r] = s2 + clb[0];
    }
}

// =====================================================================
extern "C" void kernel_launch(void* const* d_in, const int* in_sizes, int n_in,
                              void* d_out, int out_size, void* d_ws, size_t ws_size,
                              hipStream_t stream) {
    const int*   x     = (const int*)  d_in[0];
    const float* emb   = (const float*)d_in[1];
    const float* pos   = (const float*)d_in[2];
    const float* in_w  = (const float*)d_in[3];
    const float* convw = (const float*)d_in[4];
    const float* convb = (const float*)d_in[5];
    const float* dtb   = (const float*)d_in[6];
    const float* Alog  = (const float*)d_in[7];
    const float* Dpw   = (const float*)d_in[8];
    const float* normw = (const float*)d_in[9];
    const float* outw  = (const float*)d_in[10];
    const float* lnw   = (const float*)d_in[11];
    const float* lnb   = (const float*)d_in[12];
    const float* w1    = (const float*)d_in[13];
    const float* b1    = (const float*)d_in[14];
    const float* w2    = (const float*)d_in[15];
    const float* b2    = (const float*)d_in[16];
    const float* trw   = (const float*)d_in[17];
    const float* trb   = (const float*)d_in[18];
    const float* clw   = (const float*)d_in[19];
    const float* clb   = (const float*)d_in[20];
    float* out = (float*)d_out;

    // ---- workspace layout (fp32 elements) ----
    float* ws = (float*)d_ws;
    size_t off = 0;
    float* buf_h   = ws + off; off += (size_t)BL * DD;        // activations
    float* buf_zx  = ws + off; off += (size_t)BL * DIP;       // in-proj out
    float* buf_xbc = ws + off; off += (size_t)BL * CD;        // conv out
    float* buf_dt  = ws + off; off += (size_t)BL * NH;        // softplus(dt)
    float* buf_y   = ws + off; off += (size_t)BL * DI;        // scan out / normed
    float* buf_d   = ws + off; off += (size_t)BL * DD;        // out-proj out
    float* buf_mlp = ws + off; off += (size_t)BL * 4 * DD;    // mlp hidden

    k_embed<<<BL, 256, 0, stream>>>(x, emb, pos, buf_h);

    for (int i = 0; i < NLY; i++) {
        const float* in_w_i  = in_w  + (size_t)i * DIP * DD;
        const float* convw_i = convw + (size_t)i * CD * DC;
        const float* convb_i = convb + (size_t)i * CD;
        const float* dtb_i   = dtb   + (size_t)i * NH;
        const float* Alog_i  = Alog  + (size_t)i * NH;
        const float* Dp_i    = Dpw   + (size_t)i * NH;
        const float* normw_i = normw + (size_t)i * DI;
        const float* outw_i  = outw  + (size_t)i * DD * DI;
        const float* lnw_i   = lnw   + (size_t)i * DD;
        const float* lnb_i   = lnb   + (size_t)i * DD;
        const float* w1_i    = w1    + (size_t)i * 4 * DD * DD;
        const float* b1_i    = b1    + (size_t)i * 4 * DD;
        const float* w2_i    = w2    + (size_t)i * DD * 4 * DD;
        const float* b2_i    = b2    + (size_t)i * DD;

        // in-proj: (BL,256) x (1160,256)^T -> (BL,1160)
        k_gemm<<<dim3((DIP + 127) / 128, BL / 64), 256, 0, stream>>>(
            buf_h, in_w_i, buf_zx, nullptr, nullptr, BL, DIP, DD, 0);

        k_conv<<<BL, 256, 0, stream>>>(buf_zx, convw_i, convb_i, dtb_i,
                                       buf_xbc, buf_dt);

        k_scan<<<BB * NH, 256, 0, stream>>>(buf_xbc, buf_dt, Alog_i, Dp_i, buf_y);

        k_gated_norm<<<BL, 256, 0, stream>>>(buf_y, buf_zx, normw_i);

        // out-proj: (BL,512) x (256,512)^T -> (BL,256)
        k_gemm<<<dim3((DD + 127) / 128, BL / 64), 256, 0, stream>>>(
            buf_y, outw_i, buf_d, nullptr, nullptr, BL, DD, DI, 0);

        k_layernorm<<<BL, 256, 0, stream>>>(buf_d, lnw_i, lnb_i, buf_h);

        // MLP up + SiLU: (BL,256) x (1024,256)^T -> (BL,1024)
        k_gemm<<<dim3((4 * DD + 127) / 128, BL / 64), 256, 0, stream>>>(
            buf_h, w1_i, buf_mlp, b1_i, nullptr, BL, 4 * DD, DD, 1);

        // MLP down + bias + residual: writes new h
        k_gemm<<<dim3((DD + 127) / 128, BL / 64), 256, 0, stream>>>(
            buf_mlp, w2_i, buf_h, b2_i, buf_h, BL, DD, 4 * DD, 0);
    }

    k_heads<<<BL, 256, 0, stream>>>(buf_h, trw, trb, clw, clb, out);
}
